// SwinBlock_49014166782281
// MI455X (gfx1250) — compile-verified, matrix operand order checked by reading, not run on hardware
//
#include <hip/hip_runtime.h>
#include <hip/hip_bf16.h>

typedef __bf16 v16bf __attribute__((ext_vector_type(16)));
typedef float  v8f   __attribute__((ext_vector_type(8)));

#define DEVI __device__ __forceinline__

// ---------- helpers ----------
DEVI unsigned short f2bf(float f) {               // RNE f32 -> bf16 bits
    unsigned int u = __builtin_bit_cast(unsigned int, f);
    u += 0x7fffu + ((u >> 16) & 1u);
    return (unsigned short)(u >> 16);
}

// Build a 16-element bf16 WMMA fragment from two contiguous 8-element runs.
// Per CDNA5 16-bit A/B layout: elems 0..7 = K{koff..koff+7}, 8..15 = K{koff+16..+23}.
DEVI v16bf frag_ld(const unsigned short* p0, const unsigned short* p1) {
    union { uint4 v; unsigned short s[8]; } a, b;
    a.v = *reinterpret_cast<const uint4*>(p0);
    b.v = *reinterpret_cast<const uint4*>(p1);
    v16bf f;
#pragma unroll
    for (int i = 0; i < 8; ++i) {
        f[i]     = __builtin_bit_cast(__bf16, a.s[i]);
        f[i + 8] = __builtin_bit_cast(__bf16, b.s[i]);
    }
    return f;
}

DEVI float wave_sum(float v) {
#pragma unroll
    for (int off = 16; off > 0; off >>= 1) v += __shfl_xor(v, off, 32);
    return v;
}

// CDNA5 async global->LDS direct copy (16B per lane), tracked by ASYNCcnt.
// Generic pointers to __shared__ hold the LDS byte offset in their low 32 bits
DEVI void async_copy_b128(const unsigned short* gsrc, unsigned short* lds_dst) {
    unsigned lo = (unsigned)(size_t)(void*)lds_dst;
    unsigned long long ga = (unsigned long long)(size_t)(const void*)gsrc;
    asm volatile("global_load_async_to_lds_b128 %0, %1, off"
                 :: "v"(lo), "v"(ga) : "memory");
}
DEVI void async_wait0() {
    asm volatile("s_wait_asynccnt 0" ::: "memory");
}

// ---------- weight f32 -> bf16 ----------
__global__ void cvt_kernel(const float* __restrict__ src, unsigned short* __restrict__ dst, int n) {
    int i = blockIdx.x * 256 + threadIdx.x;
    if (i < n) dst[i] = f2bf(src[i]);
}

// ---------- LN1 + roll(-3,-3) + window partition (pad 49->64 rows) ----------
__global__ void __launch_bounds__(256) ln1_win_kernel(const float* __restrict__ x,
        const float* __restrict__ g, const float* __restrict__ bt,
        unsigned short* __restrict__ xwin) {
    int wv = threadIdx.x >> 5, lane = threadIdx.x & 31;
    int tok = blockIdx.x * 8 + wv;                 // 0 .. 4096*64-1
    int w = tok >> 6, n = tok & 63;
    unsigned short* dst = xwin + (size_t)tok * 128 + lane * 4;
    if (n >= 49) { dst[0] = dst[1] = dst[2] = dst[3] = 0; return; }
    int b = w >> 8, win = w & 255, wh = win >> 4, ww = win & 15;
    int r = n / 7, c = n % 7;
    int ho = (wh * 7 + r + 3) % 112;               // inverse of roll by -SHIFT
    int wo = (ww * 7 + c + 3) % 112;
    const float* src = x + ((size_t)b * 12544 + ho * 112 + wo) * 128;
    float v[4];
#pragma unroll
    for (int j = 0; j < 4; ++j) v[j] = src[lane * 4 + j];
    float s = v[0] + v[1] + v[2] + v[3];
    float sq = v[0]*v[0] + v[1]*v[1] + v[2]*v[2] + v[3]*v[3];
    s = wave_sum(s); sq = wave_sum(sq);
    float mean = s * (1.f / 128.f);
    float var  = sq * (1.f / 128.f) - mean * mean;
    float rstd = rsqrtf(var + 1e-5f);
#pragma unroll
    for (int j = 0; j < 4; ++j) {
        int cc = lane * 4 + j;
        dst[j] = f2bf((v[j] - mean) * rstd * g[cc] + bt[cc]);
    }
}

// ---------- QKV GEMM per window: (64x128) x (128x384) ----------
__global__ void __launch_bounds__(256) qkv_kernel(const unsigned short* __restrict__ xwin,
        const unsigned short* __restrict__ wq, const float* __restrict__ bq,
        unsigned short* __restrict__ qkv) {
    __shared__ __align__(16) unsigned short lx[64 * 128];
    int w = blockIdx.x;
    {
        const unsigned short* s = xwin + (size_t)w * 8192;
        for (int i = threadIdx.x; i < 1024; i += 256)
            async_copy_b128(s + i * 8, lx + i * 8);
        async_wait0();
    }
    __syncthreads();
    int wv = threadIdx.x >> 5, lane = threadIdx.x & 31;
    int lm = lane & 15, half = lane >> 4;
    for (int t = 0; t < 3; ++t) {
        int nt = wv * 3 + t;                        // 24 n-tiles over 8 waves
        for (int mt = 0; mt < 4; ++mt) {
            v8f acc = {};
#pragma unroll
            for (int kc = 0; kc < 4; ++kc) {
                int koff = kc * 32 + half * 8;
                const unsigned short* ap = lx + (mt * 16 + lm) * 128 + koff;
                const unsigned short* bp = wq + (size_t)(nt * 16 + lm) * 128 + koff;
                v16bf a = frag_ld(ap, ap + 16);
                v16bf b = frag_ld(bp, bp + 16);
                acc = __builtin_amdgcn_wmma_f32_16x16x32_bf16(false, a, false, b,
                                                              (short)0, acc, false, false);
            }
#pragma unroll
            for (int j = 0; j < 8; ++j) {
                int row = mt * 16 + j + half * 8;
                int col = nt * 16 + lm;
                float val = acc[j] + bq[col];
                int sel = col >> 7, hc = col & 127, head = hc >> 5, d = hc & 31;
                size_t idx = ((((size_t)sel * 4096 + w) * 4 + head) * 64 + row) * 32 + d;
                qkv[idx] = f2bf(val);
            }
        }
    }
}

// ---------- attention per (window, head): S=qk^T, softmax(+bias+mask), O=PV ----------
__global__ void __launch_bounds__(64) attn_kernel(const unsigned short* __restrict__ qkv,
        const float* __restrict__ mask, const float* __restrict__ bias_table,
        unsigned short* __restrict__ aout) {
    __shared__ __align__(16) float          S[2][64 * 64];
    __shared__ __align__(16) unsigned short P[2][64 * 64];
    int w = blockIdx.x;
    int wv = threadIdx.x >> 5, lane = threadIdx.x & 31;
    int h = blockIdx.y * 2 + wv;
    int lm = lane & 15, half = lane >> 4;
    const unsigned short* qb = qkv + (((size_t)0 * 4096 + w) * 4 + h) * 2048;
    const unsigned short* kb = qkv + (((size_t)1 * 4096 + w) * 4 + h) * 2048;
    const unsigned short* vb = qkv + (((size_t)2 * 4096 + w) * 4 + h) * 2048;
    float* Sh = S[wv];
    unsigned short* Ph = P[wv];

    // S = q @ k^T   (hd = 32 -> one WMMA per 16x16 tile)
    for (int mt = 0; mt < 4; ++mt)
#pragma unroll
        for (int nt = 0; nt < 4; ++nt) {
            v8f acc = {};
            int koff = half * 8;
            const unsigned short* ap = qb + (mt * 16 + lm) * 32 + koff;
            const unsigned short* bp = kb + (nt * 16 + lm) * 32 + koff;
            v16bf a = frag_ld(ap, ap + 16);
            v16bf b = frag_ld(bp, bp + 16);
            acc = __builtin_amdgcn_wmma_f32_16x16x32_bf16(false, a, false, b,
                                                          (short)0, acc, false, false);
#pragma unroll
            for (int j = 0; j < 8; ++j)
                Sh[(mt * 16 + j + half * 8) * 64 + nt * 16 + lm] = acc[j];
        }
    __syncthreads();

    // softmax rows {lane, lane+32}: scale + rel-pos bias + shift mask
    const float scale = 0.17677669529663687f;      // 1/sqrt(32)
    const float* mrow = mask + (size_t)(w & 255) * 2401;
#pragma unroll
    for (int rr = 0; rr < 2; ++rr) {
        int r = lane + rr * 32;
        float* srow = Sh + r * 64;
        unsigned short* prow = Ph + r * 64;
        if (r < 49) {
            int yr = r / 7, xr = r % 7;
            float mx = -3.0e38f;
            for (int c = 0; c < 49; ++c) {
                int yc = c / 7, xc = c % 7;
                int ridx = (yr - yc + 6) * 13 + (xr - xc + 6);
                float v = srow[c] * scale + bias_table[ridx * 4 + h] + mrow[r * 49 + c];
                srow[c] = v;
                mx = fmaxf(mx, v);
            }
            float sum = 0.f;
            for (int c = 0; c < 49; ++c) { float e = __expf(srow[c] - mx); srow[c] = e; sum += e; }
            float rinv = 1.f / sum;
            for (int c = 0; c < 49; ++c) prow[c] = f2bf(srow[c] * rinv);
            for (int c = 49; c < 64; ++c) prow[c] = 0;
        } else {
            for (int c = 0; c < 64; ++c) prow[c] = 0;
        }
    }
    __syncthreads();

    // O = P @ V   (K = 64 padded tokens -> 2 chunks)
    for (int mt = 0; mt < 4; ++mt)
#pragma unroll
        for (int nt = 0; nt < 2; ++nt) {
            v8f acc = {};
#pragma unroll
            for (int kc = 0; kc < 2; ++kc) {
                int koff = kc * 32 + half * 8;
                const unsigned short* ap = Ph + (mt * 16 + lm) * 64 + koff;
                v16bf a = frag_ld(ap, ap + 16);
                v16bf b;
                int d = nt * 16 + lm;
#pragma unroll
                for (int i = 0; i < 16; ++i) {
                    int t = koff + i + (i >= 8 ? 8 : 0);
                    b[i] = __builtin_bit_cast(__bf16, vb[t * 32 + d]);
                }
                acc = __builtin_amdgcn_wmma_f32_16x16x32_bf16(false, a, false, b,
                                                              (short)0, acc, false, false);
            }
#pragma unroll
            for (int j = 0; j < 8; ++j) {
                int row = mt * 16 + j + half * 8;
                aout[(size_t)w * 8192 + row * 128 + h * 32 + nt * 16 + lm] = f2bf(acc[j]);
            }
        }
}

// ---------- proj GEMM per window + window-reverse + roll(+3,+3) + residual ----------
__global__ void __launch_bounds__(256) proj_kernel(const unsigned short* __restrict__ aout,
        const unsigned short* __restrict__ wp, const float* __restrict__ bp,
        const float* __restrict__ xres, float* __restrict__ x1) {
    __shared__ __align__(16) unsigned short lo[64 * 128];
    int w = blockIdx.x;
    {
        const unsigned short* s = aout + (size_t)w * 8192;
        for (int i = threadIdx.x; i < 1024; i += 256)
            async_copy_b128(s + i * 8, lo + i * 8);
        async_wait0();
    }
    __syncthreads();
    int wv = threadIdx.x >> 5, lane = threadIdx.x & 31;
    int lm = lane & 15, half = lane >> 4;
    int b = w >> 8, win = w & 255, wh = win >> 4, ww = win & 15;
    int nt = wv;                                    // 8 n-tiles, one per wave
    for (int mt = 0; mt < 4; ++mt) {
        v8f acc = {};
#pragma unroll
        for (int kc = 0; kc < 4; ++kc) {
            int koff = kc * 32 + half * 8;
            const unsigned short* ap = lo + (mt * 16 + lm) * 128 + koff;
            const unsigned short* bpw = wp + (size_t)(nt * 16 + lm) * 128 + koff;
            v16bf a = frag_ld(ap, ap + 16);
            v16bf bb = frag_ld(bpw, bpw + 16);
            acc = __builtin_amdgcn_wmma_f32_16x16x32_bf16(false, a, false, bb,
                                                          (short)0, acc, false, false);
        }
#pragma unroll
        for (int j = 0; j < 8; ++j) {
            int row = mt * 16 + j + half * 8;
            if (row < 49) {
                int col = nt * 16 + lm;
                int r = row / 7, c = row % 7;
                int ho = (wh * 7 + r + 3) % 112;
                int wo = (ww * 7 + c + 3) % 112;
                size_t idx = ((size_t)b * 12544 + ho * 112 + wo) * 128 + col;
                x1[idx] = xres[idx] + acc[j] + bp[col];
            }
        }
    }
}

// ---------- LN2 ----------
__global__ void __launch_bounds__(256) ln2_kernel(const float* __restrict__ x1,
        const float* __restrict__ g, const float* __restrict__ bt,
        unsigned short* __restrict__ out) {
    int wv = threadIdx.x >> 5, lane = threadIdx.x & 31;
    int tok = blockIdx.x * 8 + wv;                 // 200704 tokens = 25088*8
    const float* src = x1 + (size_t)tok * 128;
    float v[4];
#pragma unroll
    for (int j = 0; j < 4; ++j) v[j] = src[lane * 4 + j];
    float s = v[0] + v[1] + v[2] + v[3];
    float sq = v[0]*v[0] + v[1]*v[1] + v[2]*v[2] + v[3]*v[3];
    s = wave_sum(s); sq = wave_sum(sq);
    float mean = s * (1.f / 128.f);
    float var  = sq * (1.f / 128.f) - mean * mean;
    float rstd = rsqrtf(var + 1e-5f);
    unsigned short* dst = out + (size_t)tok * 128 + lane * 4;
#pragma unroll
    for (int j = 0; j < 4; ++j) {
        int cc = lane * 4 + j;
        dst[j] = f2bf((v[j] - mean) * rstd * g[cc] + bt[cc]);
    }
}

// ---------- fc1: (200704x128) x (128x512) + GELU(exact) ----------
__global__ void __launch_bounds__(256) fc1_kernel(const unsigned short* __restrict__ a_in,
        const unsigned short* __restrict__ w1, const float* __restrict__ b1,
        unsigned short* __restrict__ out) {
    __shared__ __align__(16) unsigned short la[64 * 128];
    int rb = blockIdx.x, cb = blockIdx.y;
    {
        const unsigned short* s = a_in + (size_t)rb * 8192;
        for (int i = threadIdx.x; i < 1024; i += 256)
            async_copy_b128(s + i * 8, la + i * 8);
        async_wait0();
    }
    __syncthreads();
    int wv = threadIdx.x >> 5, lane = threadIdx.x & 31;
    int lm = lane & 15, half = lane >> 4;
    for (int s2 = 0; s2 < 2; ++s2) {
        int t = wv * 2 + s2, mt = t >> 2, nt = t & 3;
        int ncol = cb * 64 + nt * 16 + lm;
        v8f acc = {};
#pragma unroll
        for (int kc = 0; kc < 4; ++kc) {
            int koff = kc * 32 + half * 8;
            const unsigned short* ap = la + (mt * 16 + lm) * 128 + koff;
            const unsigned short* bp = w1 + (size_t)ncol * 128 + koff;
            __builtin_prefetch(bp + 128, 0, 0);
            v16bf a = frag_ld(ap, ap + 16);
            v16bf b = frag_ld(bp, bp + 16);
            acc = __builtin_amdgcn_wmma_f32_16x16x32_bf16(false, a, false, b,
                                                          (short)0, acc, false, false);
        }
#pragma unroll
        for (int j = 0; j < 8; ++j) {
            int row = rb * 64 + mt * 16 + j + half * 8;
            float v = acc[j] + b1[ncol];
            float ge = 0.5f * v * (1.0f + erff(v * 0.70710678118654752f));
            out[(size_t)row * 512 + ncol] = f2bf(ge);
        }
    }
}

// ---------- fc2: (200704x512) x (512x128) + bias + residual (in d_out) ----------
__global__ void __launch_bounds__(256) fc2_kernel(const unsigned short* __restrict__ a_in,
        const unsigned short* __restrict__ w2, const float* __restrict__ b2,
        float* __restrict__ io) {
    __shared__ __align__(16) unsigned short la[64 * 128];
    int rb = blockIdx.x, cb = blockIdx.y;
    int wv = threadIdx.x >> 5, lane = threadIdx.x & 31;
    int lm = lane & 15, half = lane >> 4;
    int t0 = wv * 2, t1 = wv * 2 + 1;
    int mt0 = t0 >> 2, nt0 = t0 & 3, mt1 = t1 >> 2, nt1 = t1 & 3;
    v8f acc0 = {}, acc1 = {};
    for (int ks = 0; ks < 4; ++ks) {
        if (ks) __syncthreads();
        for (int i = threadIdx.x; i < 1024; i += 256) {
            int row = i >> 4, q = i & 15;
            const unsigned short* s =
                a_in + (size_t)(rb * 64 + row) * 512 + ks * 128 + q * 8;
            async_copy_b128(s, la + i * 8);
        }
        async_wait0();
        __syncthreads();
#pragma unroll
        for (int kc = 0; kc < 4; ++kc) {
            int koff = kc * 32 + half * 8;
            int kg = ks * 128 + koff;
            const unsigned short* ap0 = la + (mt0 * 16 + lm) * 128 + koff;
            const unsigned short* bp0 = w2 + (size_t)(cb * 64 + nt0 * 16 + lm) * 512 + kg;
            v16bf a0 = frag_ld(ap0, ap0 + 16);
            v16bf b0 = frag_ld(bp0, bp0 + 16);
            acc0 = __builtin_amdgcn_wmma_f32_16x16x32_bf16(false, a0, false, b0,
                                                           (short)0, acc0, false, false);
            const unsigned short* ap1 = la + (mt1 * 16 + lm) * 128 + koff;
            const unsigned short* bp1 = w2 + (size_t)(cb * 64 + nt1 * 16 + lm) * 512 + kg;
            v16bf a1 = frag_ld(ap1, ap1 + 16);
            v16bf b1v = frag_ld(bp1, bp1 + 16);
            acc1 = __builtin_amdgcn_wmma_f32_16x16x32_bf16(false, a1, false, b1v,
                                                           (short)0, acc1, false, false);
        }
    }
#pragma unroll
    for (int j = 0; j < 8; ++j) {
        int row0 = rb * 64 + mt0 * 16 + j + half * 8;
        int col0 = cb * 64 + nt0 * 16 + lm;
        size_t i0 = (size_t)row0 * 128 + col0;
        io[i0] = io[i0] + acc0[j] + b2[col0];
        int row1 = rb * 64 + mt1 * 16 + j + half * 8;
        int col1 = cb * 64 + nt1 * 16 + lm;
        size_t i1 = (size_t)row1 * 128 + col1;
        io[i1] = io[i1] + acc1[j] + b2[col1];
    }
}

// ---------- host launcher ----------
extern "C" void kernel_launch(void* const* d_in, const int* in_sizes, int n_in,
                              void* d_out, int out_size, void* d_ws, size_t ws_size,
                              hipStream_t stream) {
    (void)in_sizes; (void)n_in; (void)out_size; (void)ws_size;
    const float* x      = (const float*)d_in[0];
    const float* mask   = (const float*)d_in[3];
    const float* ln1_g  = (const float*)d_in[4];
    const float* ln1_b  = (const float*)d_in[5];
    const float* qkv_w  = (const float*)d_in[6];
    const float* qkv_b  = (const float*)d_in[7];
    const float* relb   = (const float*)d_in[8];
    const float* proj_w = (const float*)d_in[9];
    const float* proj_b = (const float*)d_in[10];
    const float* ln2_g  = (const float*)d_in[11];
    const float* ln2_b  = (const float*)d_in[12];
    const float* fc1_w  = (const float*)d_in[13];
    const float* fc1_b  = (const float*)d_in[14];
    const float* fc2_w  = (const float*)d_in[15];
    const float* fc2_b  = (const float*)d_in[16];
    float* out = (float*)d_out;

    char* ws = (char*)d_ws;
    size_t off = 0;
    auto alloc = [&](size_t bytes) {
        void* p = ws + off;
        off += (bytes + 255) & ~(size_t)255;
        return p;
    };
    unsigned short* wqkv  = (unsigned short*)alloc((size_t)49152 * 2);
    unsigned short* wproj = (unsigned short*)alloc((size_t)16384 * 2);
    unsigned short* wfc1  = (unsigned short*)alloc((size_t)65536 * 2);
    unsigned short* wfc2  = (unsigned short*)alloc((size_t)65536 * 2);
    unsigned short* xwin  = (unsigned short*)alloc((size_t)4096 * 8192 * 2);
    unsigned short* qkvb  = (unsigned short*)alloc((size_t)3 * 4096 * 4 * 64 * 32 * 2);
    unsigned short* aout  = (unsigned short*)alloc((size_t)4096 * 8192 * 2);
    unsigned short* l2o   = (unsigned short*)alloc((size_t)200704 * 128 * 2);
    unsigned short* f1o   = (unsigned short*)alloc((size_t)200704 * 512 * 2);

    cvt_kernel<<<192, 256, 0, stream>>>(qkv_w, wqkv, 49152);
    cvt_kernel<<<64, 256, 0, stream>>>(proj_w, wproj, 16384);
    cvt_kernel<<<256, 256, 0, stream>>>(fc1_w, wfc1, 65536);
    cvt_kernel<<<256, 256, 0, stream>>>(fc2_w, wfc2, 65536);

    ln1_win_kernel<<<32768, 256, 0, stream>>>(x, ln1_g, ln1_b, xwin);
    qkv_kernel<<<4096, 256, 0, stream>>>(xwin, wqkv, qkv_b, qkvb);
    attn_kernel<<<dim3(4096, 2), 64, 0, stream>>>(qkvb, mask, relb, aout);
    proj_kernel<<<4096, 256, 0, stream>>>(aout, wproj, proj_b, x, out);
    ln2_kernel<<<25088, 256, 0, stream>>>(out, ln2_g, ln2_b, l2o);
    fc1_kernel<<<dim3(3136, 8), 256, 0, stream>>>(l2o, wfc1, fc1_b, f1o);
    fc2_kernel<<<dim3(3136, 2), 256, 0, stream>>>(f1o, wfc2, fc2_b, out);
}
